// OptimizedAttention_13932873908622
// MI455X (gfx1250) — matrix operand order repaired and checked by reading, not compiled, not run on hardware
//
#include <hip/hip_runtime.h>

typedef __attribute__((ext_vector_type(16))) _Float16 v16h;
typedef __attribute__((ext_vector_type(8)))  float    v8f;

#define NUM_HEADS    16
#define NUM_KV_HEADS 4
#define SEQ_LEN      2048
#define HEAD_DIM     64
#define N_REP        4
#define QTILE        64     // query rows per workgroup (4 waves x 16)
#define KVTILE       32     // kv rows per pipeline step
#define NSTEPS       (SEQ_LEN / KVTILE)
#define KSTRIDE      80     // halves per row for Q/K LDS tiles (160B rows, 32B-aligned chunks)
#define VSTRIDE      48     // halves per row for transposed V LDS (96B rows)
#define PSTRIDE      40     // halves per row for per-wave P staging (80B rows)

// fold softmax scale (1/sqrt(64)) and log2(e) into Q: scores land in the
// log2 domain, so softmax exponentials are single v_exp_f32 ops.
#define QSCALE (0.125f * 1.44269504088896340736f)

// 16-lane butterfly max using DPP (pure VALU, no LDS traffic).
__device__ __forceinline__ float dpp_max16(float v) {
    int x;
    x = __builtin_amdgcn_update_dpp(0, __float_as_int(v), 0xB1,  0xf, 0xf, true); // quad_perm [1,0,3,2]
    v = fmaxf(v, __int_as_float(x));
    x = __builtin_amdgcn_update_dpp(0, __float_as_int(v), 0x4E,  0xf, 0xf, true); // quad_perm [2,3,0,1]
    v = fmaxf(v, __int_as_float(x));
    x = __builtin_amdgcn_update_dpp(0, __float_as_int(v), 0x141, 0xf, 0xf, true); // row_half_mirror
    v = fmaxf(v, __int_as_float(x));
    x = __builtin_amdgcn_update_dpp(0, __float_as_int(v), 0x140, 0xf, 0xf, true); // row_mirror
    v = fmaxf(v, __int_as_float(x));
    return v;
}

__global__ __launch_bounds__(128, 1)
void fa_rope_gqa_wmma(const float* __restrict__ Q, const float* __restrict__ K,
                      const float* __restrict__ V, const float* __restrict__ COS,
                      const float* __restrict__ SIN, const int* __restrict__ POS,
                      float* __restrict__ OUT)
{
    __shared__ _Float16 q_lds[QTILE * KSTRIDE];            // 10240 B (RoPE'd, pre-scaled Q)
    __shared__ _Float16 k_lds[2][KVTILE * KSTRIDE];        // 2 x 5120 B (double-buffered K)
    __shared__ _Float16 vT_lds[2][HEAD_DIM * VSTRIDE];     // 2 x 6144 B (double-buffered V^T, rows permuted)
    __shared__ _Float16 p_lds[4][16 * PSTRIDE];            // 5120 B (per-wave P staging, cols interleaved)

    const int tid  = threadIdx.x;
    const int lane = tid & 31;
    const int wave = tid >> 5;
    const int g    = lane >> 4;     // half-wave group (0/1)
    const int lo   = lane & 15;     // lane within half

    const int head  = blockIdx.x >> 5;      // 32 q-tiles per head
    const int qt    = blockIdx.x & 31;
    const int qbase = qt * QTILE;
    const int kvh   = head / N_REP;

    const float* qh = Q + (size_t)head * SEQ_LEN * HEAD_DIM;
    const float* kh = K + (size_t)kvh  * SEQ_LEN * HEAD_DIM;
    const float* vh = V + (size_t)kvh  * SEQ_LEN * HEAD_DIM;

    // ---- Q tile: RoPE, fold in scale*log2e -> f16 LDS ---------------------
    #pragma unroll
    for (int i = 0; i < (QTILE * 32) / 128; ++i) {
        int idx = tid + i * 128;
        int r  = idx >> 5;
        int dp = idx & 31;
        int s  = qbase + r;
        int pos = POS[s];
        float c  = COS[pos * HEAD_DIM + dp];
        float sn = SIN[pos * HEAD_DIM + dp];
        float x0 = qh[(size_t)s * HEAD_DIM + dp];
        float x1 = qh[(size_t)s * HEAD_DIM + dp + 32];
        q_lds[r * KSTRIDE + dp]      = (_Float16)((x0 * c - x1 * sn) * QSCALE);
        q_lds[r * KSTRIDE + dp + 32] = (_Float16)((x1 * c + x0 * sn) * QSCALE);
    }

    // ---- staging registers for software-pipelined K/V fetch ---------------
    float kx0[8], kx1[8], kc[8], ks[8];
    float vx[16];

    auto stage_load = [&](int tbase) {
        #pragma unroll
        for (int i = 0; i < 8; ++i) {           // K: 32 rows x 32 pairs / 128 thr
            int idx = tid + i * 128;
            int r  = idx >> 5;
            int dp = idx & 31;
            int t  = tbase + r;
            int pos = POS[t];
            kc[i]  = COS[pos * HEAD_DIM + dp];
            ks[i]  = SIN[pos * HEAD_DIM + dp];
            kx0[i] = kh[(size_t)t * HEAD_DIM + dp];
            kx1[i] = kh[(size_t)t * HEAD_DIM + dp + 32];
        }
        #pragma unroll
        for (int i = 0; i < 16; ++i) {          // V: 32 rows x 64 / 128 thr
            int idx = tid + i * 128;
            int r = idx >> 6;
            int d = idx & 63;
            vx[i] = vh[(size_t)(tbase + r) * HEAD_DIM + d];
        }
    };
    auto stage_store = [&](int buf) {
        _Float16* kb = k_lds[buf];
        _Float16* vb = vT_lds[buf];
        #pragma unroll
        for (int i = 0; i < 8; ++i) {
            int idx = tid + i * 128;
            int r  = idx >> 5;
            int dp = idx & 31;
            kb[r * KSTRIDE + dp]      = (_Float16)(kx0[i] * kc[i] - kx1[i] * ks[i]);
            kb[r * KSTRIDE + dp + 32] = (_Float16)(kx1[i] * kc[i] + kx0[i] * ks[i]);
        }
        #pragma unroll
        for (int i = 0; i < 16; ++i) {
            int idx = tid + i * 128;
            int r = idx >> 6;
            int d = idx & 63;
            // permute kv rows to match interleaved P columns: t -> 2*(t&15) + (t>>4)
            int rp = ((r & 15) << 1) | (r >> 4);
            vb[d * VSTRIDE + rp] = (_Float16)vx[i];   // transposed + permuted store
        }
    };

    stage_load(0);          // issue first tile's global loads early
    __syncthreads();        // q_lds ready

    // ---- gather per-wave Q as two 16x32 f16 A-operands (contiguous k = 16*g + h, same order as B) -----
    v16h aq0, aq1;
    {
        const int qrow = wave * 16 + lo;
        #pragma unroll
        for (int h = 0; h < 16; ++h) {
            int kl = g * 16 + h;
            aq0[h] = q_lds[qrow * KSTRIDE + kl];
            aq1[h] = q_lds[qrow * KSTRIDE + 32 + kl];
        }
    }

    const v8f vzero = {0.f, 0.f, 0.f, 0.f, 0.f, 0.f, 0.f, 0.f};
    v8f acc[4];
    acc[0] = vzero; acc[1] = vzero; acc[2] = vzero; acc[3] = vzero;

    v16h ones;
    #pragma unroll
    for (int h = 0; h < 16; ++h) ones[h] = (_Float16)1.0f;

    float mrow[8], lrow[8];
    #pragma unroll
    for (int r = 0; r < 8; ++r) { mrow[r] = -1.0e30f; lrow[r] = 0.0f; }

    _Float16* pw = p_lds[wave];

    // ======================= pipelined KV sweep ============================
    for (int kt = 0; kt < NSTEPS; ++kt) {
        const int buf = kt & 1;
        stage_store(buf);               // commit tile kt (staged regs -> LDS)
        __syncthreads();                // tile kt visible; prev compute done
        if (kt + 1 < NSTEPS) stage_load((kt + 1) * KVTILE);   // overlap next fetch
        if (kt + 2 < NSTEPS) {          // warm L2 two tiles ahead
            const char* kn = (const char*)(kh + (size_t)(kt + 2) * KVTILE * HEAD_DIM);
            const char* vn = (const char*)(vh + (size_t)(kt + 2) * KVTILE * HEAD_DIM);
            __builtin_prefetch(kn + tid * 64, 0, 1);
            __builtin_prefetch(vn + tid * 64, 0, 1);
        }

        const _Float16* kb = k_lds[buf];
        const _Float16* vb = vT_lds[buf];

        // ---- S = Q K^T (log2 domain); two 16x16 score tiles ---------------
        v8f c0 = vzero, c1 = vzero;
        {
            v16h b00 = *(const v16h*)&kb[(lo)      * KSTRIDE +      g * 16];
            v16h b01 = *(const v16h*)&kb[(lo)      * KSTRIDE + 32 + g * 16];
            v16h b10 = *(const v16h*)&kb[(16 + lo) * KSTRIDE +      g * 16];
            v16h b11 = *(const v16h*)&kb[(16 + lo) * KSTRIDE + 32 + g * 16];
            c0 = __builtin_amdgcn_wmma_f32_16x16x32_f16(false, aq0, false, b00, (short)0, c0, false, false);
            c0 = __builtin_amdgcn_wmma_f32_16x16x32_f16(false, aq1, false, b01, (short)0, c0, false, false);
            c1 = __builtin_amdgcn_wmma_f32_16x16x32_f16(false, aq0, false, b10, (short)0, c1, false, false);
            c1 = __builtin_amdgcn_wmma_f32_16x16x32_f16(false, aq1, false, b11, (short)0, c1, false, false);
        }

        // ---- online softmax: DPP max, exp2, packed-b32 P staging ----------
        float alpha[8];
        #pragma unroll
        for (int r = 0; r < 8; ++r) {
            float s0 = c0[r];
            float s1 = c1[r];
            float mt = dpp_max16(fmaxf(s0, s1));
            float nm = fmaxf(mrow[r], mt);
            float p0 = __builtin_amdgcn_exp2f(s0 - nm);
            float p1 = __builtin_amdgcn_exp2f(s1 - nm);
            float a  = __builtin_amdgcn_exp2f(mrow[r] - nm);
            mrow[r]  = nm;
            lrow[r] *= a;
            alpha[r] = a;
            // interleaved columns: lane's (p0,p1) pair is contiguous -> 1 b32 store
            union { _Float16 h[2]; unsigned int u; } pk;
            pk.h[0] = (_Float16)p0;
            pk.h[1] = (_Float16)p1;
            int m = g * 8 + r;
            *(unsigned int*)&pw[m * PSTRIDE + 2 * lo] = pk.u;
        }

        // rescale output accumulators by alpha per row
        #pragma unroll
        for (int ch = 0; ch < 4; ++ch)
            #pragma unroll
            for (int r = 0; r < 8; ++r)
                acc[ch][r] *= alpha[r];

        // ---- regather P as 16x32 f16 A-operand ----------------------------
        v16h ap;
        #pragma unroll
        for (int h = 0; h < 16; ++h) {
            int kl = g * 16 + h;
            ap[h] = pw[lo * PSTRIDE + kl];
        }

        // ---- row sums via WMMA against all-ones B -------------------------
        v8f csum = __builtin_amdgcn_wmma_f32_16x16x32_f16(false, ap, false, ones, (short)0, vzero, false, false);
        #pragma unroll
        for (int r = 0; r < 8; ++r) lrow[r] += csum[r];

        // ---- O += P V  (4 N-chunks of 16 columns) -------------------------
        #pragma unroll
        for (int ch = 0; ch < 4; ++ch) {
            v16h bv = *(const v16h*)&vb[(ch * 16 + lo) * VSTRIDE + g * 16];
            acc[ch] = __builtin_amdgcn_wmma_f32_16x16x32_f16(false, ap, false, bv, (short)0, acc[ch], false, false);
        }
    }

    // ---- finalize: divide by row sums, store fp32 -------------------------
    float inv[8];
    #pragma unroll
    for (int r = 0; r < 8; ++r) inv[r] = 1.0f / lrow[r];
    #pragma unroll
    for (int ch = 0; ch < 4; ++ch) {
        #pragma unroll
        for (int r = 0; r < 8; ++r) {
            int s = qbase + wave * 16 + g * 8 + r;
            OUT[((size_t)head * SEQ_LEN + s) * HEAD_DIM + ch * 16 + lo] = acc[ch][r] * inv[r];
        }
    }
}

extern "C" void kernel_launch(void* const* d_in, const int* in_sizes, int n_in,
                              void* d_out, int out_size, void* d_ws, size_t ws_size,
                              hipStream_t stream) {
    const float* q   = (const float*)d_in[0];
    const float* k   = (const float*)d_in[1];
    const float* v   = (const float*)d_in[2];
    const float* cos = (const float*)d_in[3];
    const float* sin = (const float*)d_in[4];
    const int*   pos = (const int*)  d_in[5];
    float* out = (float*)d_out;

    dim3 grid(NUM_HEADS * (SEQ_LEN / QTILE));   // 16 * 32 = 512 workgroups
    dim3 block(128);                             // 4 wave32s
    fa_rope_gqa_wmma<<<grid, block, 0, stream>>>(q, k, v, cos, sin, pos, out);
}
